// EncoderDecoder_36532991820510
// MI455X (gfx1250) — matrix operand order repaired and checked by reading, not compiled, not run on hardware
//
#include <hip/hip_runtime.h>

// ---------------------------------------------------------------------------
// CDNA5 (gfx1250, wave32) implementation of the encoder/decoder + bcim model.
// GEMMs/attention on v_wmma_f32_16x16x32_f16; LDS tiles staged with
// GLOBAL_LOAD_ASYNC_TO_LDS_B128 (ASYNCcnt) and double buffering.
// ---------------------------------------------------------------------------

typedef _Float16 f16;
typedef __attribute__((ext_vector_type(16))) _Float16 v16h;
typedef __attribute__((ext_vector_type(8)))  _Float16 h8;
typedef __attribute__((ext_vector_type(8)))  float    v8f;

static constexpr int BATCH  = 4;
static constexpr int SEQ    = 2048;
static constexpr int DMODEL = 768;
static constexpr int HDIM   = 64;
static constexpr int FFN    = 3072;

#define CAT16(lo, hi) __builtin_shufflevector((lo), (hi), 0,1,2,3,4,5,6,7,8,9,10,11,12,13,14,15)
#define WMMA_F16(a, b, c) __builtin_amdgcn_wmma_f32_16x16x32_f16(false, (a), false, (b), (short)0, (c), false, false)

// ---------------------------------------------------------------------------
// Async global->LDS copy (16 bytes/lane). VDST holds the LDS byte address
// (low 32 bits of the generic shared pointer).
// Tracked by ASYNCcnt; we wait explicitly before barriers.
// ---------------------------------------------------------------------------
#define USE_ASYNC_LDS 1

__device__ __forceinline__ void async_cp16(const f16* gsrc, f16* ldst) {
#if USE_ASYNC_LDS
  const unsigned l32 = (unsigned)(unsigned long long)ldst;
  asm volatile("global_load_async_to_lds_b128 %0, %1, off"
               :: "v"(l32), "v"(gsrc) : "memory");
#else
  const uint4* s = (const uint4*)gsrc;
  uint4* d = (uint4*)ldst;
  d[0] = s[0];
#endif
}

__device__ __forceinline__ void wait_async0() {
#if USE_ASYNC_LDS
#if __has_builtin(__builtin_amdgcn_s_wait_asynccnt)
  __builtin_amdgcn_s_wait_asynccnt(0);
#else
  asm volatile("s_wait_asynccnt 0" ::: "memory");
#endif
#endif
}

// ---------------------------------------------------------------------------
// Block-wide sum reduction (256 threads = 8 wave32 waves)
// ---------------------------------------------------------------------------
__device__ __forceinline__ float block_reduce_sum(float v, float* sm) {
  #pragma unroll
  for (int o = 16; o > 0; o >>= 1) v += __shfl_down(v, o, 32);
  const int wid = threadIdx.x >> 5, lane = threadIdx.x & 31;
  if (lane == 0) sm[wid] = v;
  __syncthreads();
  if (wid == 0) {
    v = (lane < 8) ? sm[lane] : 0.0f;
    #pragma unroll
    for (int o = 4; o > 0; o >>= 1) v += __shfl_down(v, o, 32);
    if (lane == 0) sm[0] = v;
  }
  __syncthreads();
  v = sm[0];
  __syncthreads();
  return v;
}

// ---------------------------------------------------------------------------
// LayerNorm over D=768: one block per row, writes f16 (and optional f32).
// ---------------------------------------------------------------------------
__global__ __launch_bounds__(256) void ln_kernel(const float* __restrict__ src,
                                                 const float* __restrict__ gma,
                                                 const float* __restrict__ bta,
                                                 f16* __restrict__ o16,
                                                 float* __restrict__ o32) {
  __shared__ float sm[8];
  const int row = blockIdx.x;
  const float* x = src + (size_t)row * DMODEL;
  float v[3];
  #pragma unroll
  for (int i = 0; i < 3; i++) v[i] = x[threadIdx.x + i * 256];
  float s = v[0] + v[1] + v[2];
  s = block_reduce_sum(s, sm);
  const float mean = s * (1.0f / DMODEL);
  float q = 0.0f;
  #pragma unroll
  for (int i = 0; i < 3; i++) { const float d = v[i] - mean; q += d * d; }
  q = block_reduce_sum(q, sm);
  const float rstd = rsqrtf(q * (1.0f / DMODEL) + 1e-5f);
  #pragma unroll
  for (int i = 0; i < 3; i++) {
    const int c = threadIdx.x + i * 256;
    const float y = (v[i] - mean) * rstd * gma[c] + bta[c];
    o16[(size_t)row * DMODEL + c] = (f16)y;
    if (o32) o32[(size_t)row * DMODEL + c] = y;
  }
}

// ---------------------------------------------------------------------------
// Weight convert+transpose: W f32 [K,N] -> Wt f16 [N,K]
// ---------------------------------------------------------------------------
__global__ void transpose_f16(const float* __restrict__ W, f16* __restrict__ Wt,
                              int K, int N) {
  const int idx = blockIdx.x * 256 + threadIdx.x;
  if (idx < K * N) {
    const int k = idx / N, n = idx - k * N;
    Wt[(size_t)n * K + k] = (f16)W[idx];
  }
}

// ---------------------------------------------------------------------------
// WMMA GEMM: C[M,N] = act((A[M,K] @ Bt[N,K]^T + bias) * scale) + res
//   - 8 waves, tile 128x128, K-step 32, double-buffered async-staged LDS
//   - per-wave 64x32 subtile -> 4x2 accumulator fragments
// ---------------------------------------------------------------------------
#define GSTR 40  // padded LDS row stride (halves): conflict-free, 16B aligned

__global__ __launch_bounds__(256) void gemm_wmma(const f16* __restrict__ A,
                                                 const f16* __restrict__ Bt,
                                                 const float* __restrict__ bias,
                                                 const float* __restrict__ res,
                                                 unsigned resMask,
                                                 float* __restrict__ o32,
                                                 f16* __restrict__ o16,
                                                 int N, int K, float scale,
                                                 int act_gelu) {
  __shared__ __align__(16) f16 As[2][128 * GSTR];
  __shared__ __align__(16) f16 Bs[2][128 * GSTR];

  const int t = threadIdx.x;
  const int lane = t & 31, wave = t >> 5;
  const int lane15 = lane & 15, hs = lane >> 4;
  const int wm = wave & 1, wn = wave >> 1;
  const int m0 = blockIdx.y * 128, n0 = blockIdx.x * 128;
  const int srow = t >> 1;           // 0..127 staging row
  const int sseg = (t & 1) << 4;     // 0 or 16 halves

  const f16* aRow = A + (size_t)(m0 + srow) * K + sseg;
  const f16* bRow = Bt + (size_t)(n0 + srow) * K + sseg;

  auto stage = [&](int kk, int buf) {
    f16* ad = &As[buf][srow * GSTR + sseg];
    f16* bd = &Bs[buf][srow * GSTR + sseg];
#if USE_ASYNC_LDS
    async_cp16(aRow + kk, ad);
    async_cp16(aRow + kk + 8, ad + 8);
    async_cp16(bRow + kk, bd);
    async_cp16(bRow + kk + 8, bd + 8);
#else
    const uint4* asrc = (const uint4*)(aRow + kk);
    ((uint4*)ad)[0] = asrc[0]; ((uint4*)ad)[1] = asrc[1];
    const uint4* bsrc = (const uint4*)(bRow + kk);
    ((uint4*)bd)[0] = bsrc[0]; ((uint4*)bd)[1] = bsrc[1];
#endif
  };

  v8f acc[4][2] = {};

  const int nk = K >> 5;
  stage(0, 0);
  wait_async0();
  __syncthreads();

  for (int kt = 0; kt < nk; kt++) {
    const int cur = kt & 1;
    if (kt + 1 < nk) stage((kt + 1) << 5, cur ^ 1);  // DMA next tile during compute

    const f16* asb = As[cur];
    const f16* bsb = Bs[cur];
    v16h af[4], bf[2];
    #pragma unroll
    for (int mf = 0; mf < 4; mf++) {
      const f16* p = asb + (wm * 64 + mf * 16 + lane15) * GSTR;
      h8 lo = *(const h8*)(p + hs * 8);
      h8 hi = *(const h8*)(p + 16 + hs * 8);
      af[mf] = CAT16(lo, hi);
    }
    #pragma unroll
    for (int nf = 0; nf < 2; nf++) {
      const f16* p = bsb + (wn * 32 + nf * 16 + lane15) * GSTR;
      h8 lo = *(const h8*)(p + hs * 8);
      h8 hi = *(const h8*)(p + 16 + hs * 8);
      bf[nf] = CAT16(lo, hi);
    }
    #pragma unroll
    for (int mf = 0; mf < 4; mf++)
      #pragma unroll
      for (int nf = 0; nf < 2; nf++)
        acc[mf][nf] = WMMA_F16(af[mf], bf[nf], acc[mf][nf]);

    if (kt + 1 < nk) wait_async0();
    __syncthreads();
  }

  #pragma unroll
  for (int mf = 0; mf < 4; mf++) {
    #pragma unroll
    for (int nf = 0; nf < 2; nf++) {
      const int gn = n0 + wn * 32 + nf * 16 + lane15;
      const float bv = bias ? bias[gn] : 0.0f;
      #pragma unroll
      for (int r = 0; r < 8; r++) {
        const int gm = m0 + wm * 64 + mf * 16 + r + 8 * hs;
        float v = (acc[mf][nf][r] + bv) * scale;
        if (act_gelu) v = 0.5f * v * (1.0f + erff(v * 0.70710678118654752f));
        if (res) v += res[(size_t)((unsigned)gm & resMask) * N + gn];
        if (o32) o32[(size_t)gm * N + gn] = v;
        if (o16) o16[(size_t)gm * N + gn] = (f16)v;
      }
    }
  }
}

// ---------------------------------------------------------------------------
// Flash attention: softmax(Q K^T) V, per (qtile=128, head, batch) workgroup.
// Q scale (1/8) pre-folded into the Q projection GEMM.
// Each wave owns 16 q-rows; 64-key tiles; K async-DMA'd to LDS once per WG,
// V transposed into LDS for the PV B-fragments.
// ---------------------------------------------------------------------------
__global__ __launch_bounds__(256) void attn_kernel(const f16* __restrict__ Q,
                                                   long qBatchStride,
                                                   const f16* __restrict__ Kg,
                                                   const f16* __restrict__ Vg,
                                                   f16* __restrict__ Og) {
  __shared__ __align__(16) f16 ks[64 * 72];        // K tile:   [key][d]
  __shared__ __align__(16) f16 vt[64 * 72];        // V^T tile: [d][key]
  __shared__ __align__(16) f16 ph[8 * 16 * 64];    // per-wave prob scratch

  const int t = threadIdx.x, lane = t & 31, wave = t >> 5;
  const int lane15 = lane & 15, hs = lane >> 4;
  const int head = blockIdx.y, b = blockIdx.z;
  const int qq0 = blockIdx.x * 128 + wave * 16;

  // Q fragments (A-layout), 2 d-chunks of 32
  v16h qf[2];
  #pragma unroll
  for (int dc = 0; dc < 2; dc++) {
    const f16* p = Q + (size_t)qBatchStride * b +
                   (size_t)(qq0 + lane15) * DMODEL + head * HDIM + dc * 32 + hs * 8;
    h8 lo = *(const h8*)p;
    h8 hi = *(const h8*)(p + 16);
    qf[dc] = CAT16(lo, hi);
  }

  v8f Oc[4] = {};
  float mrun[8], lrun[8];
  #pragma unroll
  for (int r = 0; r < 8; r++) { mrun[r] = -3.0e38f; lrun[r] = 0.0f; }

  const int vkk = t & 63, vdb = (t >> 6) * 16;    // V staging assignment
  const int krow = t >> 2, kseg = (t & 3) * 16;   // K staging assignment

  for (int k0 = 0; k0 < SEQ; k0 += 64) {
    // K tile via async DMA (row copy)
    {
      const f16* kp = Kg + ((size_t)b * SEQ + k0 + krow) * DMODEL + head * HDIM + kseg;
      f16* kd = &ks[krow * 72 + kseg];
#if USE_ASYNC_LDS
      async_cp16(kp, kd);
      async_cp16(kp + 8, kd + 8);
#else
      ((uint4*)kd)[0] = ((const uint4*)kp)[0];
      ((uint4*)kd)[1] = ((const uint4*)kp)[1];
#endif
    }
    // V tile transposed (VALU scatter; no transposing DMA)
    {
      const f16* vp = Vg + ((size_t)b * SEQ + k0 + vkk) * DMODEL + head * HDIM + vdb;
      h8 va = *(const h8*)vp;
      h8 vb = *(const h8*)(vp + 8);
      #pragma unroll
      for (int i = 0; i < 8; i++) {
        vt[(vdb + i) * 72 + vkk]     = va[i];
        vt[(vdb + 8 + i) * 72 + vkk] = vb[i];
      }
    }
    wait_async0();
    __syncthreads();

    // S = Q K^T  (B-fragments: K rows = columns of K^T, from LDS)
    v8f sa[4] = {};
    #pragma unroll
    for (int dc = 0; dc < 2; dc++) {
      #pragma unroll
      for (int nf = 0; nf < 4; nf++) {
        const f16* kp = ks + (nf * 16 + lane15) * 72 + dc * 32;
        h8 lo = *(const h8*)(kp + hs * 8);
        h8 hi = *(const h8*)(kp + 16 + hs * 8);
        v16h kf = CAT16(lo, hi);
        sa[nf] = WMMA_F16(qf[dc], kf, sa[nf]);
      }
    }

    // online softmax: each 16-lane half-wave owns a fixed set of 8 rows
    #pragma unroll
    for (int r = 0; r < 8; r++) {
      float mt = fmaxf(fmaxf(sa[0][r], sa[1][r]), fmaxf(sa[2][r], sa[3][r]));
      #pragma unroll
      for (int msk = 1; msk < 16; msk <<= 1) mt = fmaxf(mt, __shfl_xor(mt, msk, 16));
      const float mnew = fmaxf(mrun[r], mt);
      const float alpha = __expf(mrun[r] - mnew);
      mrun[r] = mnew;
      float rs = 0.0f;
      #pragma unroll
      for (int nf = 0; nf < 4; nf++) {
        const float pv = __expf(sa[nf][r] - mnew);
        sa[nf][r] = pv;
        rs += pv;
      }
      #pragma unroll
      for (int msk = 1; msk < 16; msk <<= 1) rs += __shfl_xor(rs, msk, 16);
      lrun[r] = lrun[r] * alpha + rs;
      #pragma unroll
      for (int nf = 0; nf < 4; nf++) Oc[nf][r] *= alpha;
    }

    // C-frag -> A-frag relayout of probabilities via wave-private LDS
    #pragma unroll
    for (int nf = 0; nf < 4; nf++)
      #pragma unroll
      for (int r = 0; r < 8; r++)
        ph[(wave * 16 + r + 8 * hs) * 64 + nf * 16 + lane15] = (f16)sa[nf][r];

    // O += P V
    #pragma unroll
    for (int kc = 0; kc < 2; kc++) {
      const f16* pp = ph + (wave * 16 + lane15) * 64 + kc * 32 + hs * 8;
      h8 plo = *(const h8*)pp;
      h8 phi = *(const h8*)(pp + 16);
      v16h pf = CAT16(plo, phi);
      #pragma unroll
      for (int nf = 0; nf < 4; nf++) {
        const f16* vpp = vt + (nf * 16 + lane15) * 72 + kc * 32;
        h8 vlo = *(const h8*)(vpp + hs * 8);
        h8 vhi = *(const h8*)(vpp + 16 + hs * 8);
        v16h vf = CAT16(vlo, vhi);
        Oc[nf] = WMMA_F16(pf, vf, Oc[nf]);
      }
    }
    __syncthreads();
  }

  #pragma unroll
  for (int nf = 0; nf < 4; nf++)
    #pragma unroll
    for (int r = 0; r < 8; r++) {
      const int gq = qq0 + r + 8 * hs;
      Og[((size_t)b * SEQ + gq) * DMODEL + head * HDIM + nf * 16 + lane15] =
          (f16)(Oc[nf][r] / lrun[r]);
    }
}

// ---------------------------------------------------------------------------
// bcim: channel norms, 3x3 cosine similarity, scale.
// Spatial position sp in [0,1024); channels = x[b, sp, :] ++ x[b, sp+1024, :].
// ---------------------------------------------------------------------------
__global__ __launch_bounds__(256) void norm_kernel(const float* __restrict__ x,
                                                   float* __restrict__ norms) {
  __shared__ float sm[8];
  const int b = blockIdx.x >> 10, sp = blockIdx.x & 1023;
  const float* r0 = x + ((size_t)b * SEQ + sp) * DMODEL;
  const float* r1 = x + ((size_t)b * SEQ + 1024 + sp) * DMODEL;
  float s = 0.0f;
  for (int i = threadIdx.x; i < DMODEL; i += 256) {
    const float a = r0[i], c = r1[i];
    s += a * a + c * c;
  }
  s = block_reduce_sum(s, sm);
  if (threadIdx.x == 0) norms[blockIdx.x] = sqrtf(s);
}

__global__ __launch_bounds__(256) void sim_kernel(const float* __restrict__ x,
                                                  const float* __restrict__ norms,
                                                  float* __restrict__ sim) {
  __shared__ float cu[1536];
  __shared__ float sm[8];
  const int b = blockIdx.x >> 10, sp = blockIdx.x & 1023;
  const int hh = sp >> 5, ww = sp & 31;
  for (int i = threadIdx.x; i < 1536; i += 256) {
    const int hf = (i >= DMODEL) ? 1 : 0;
    const int c = i - hf * DMODEL;
    cu[i] = x[((size_t)b * SEQ + hf * 1024 + sp) * DMODEL + c];
  }
  __syncthreads();
  const float cn = norms[b * 1024 + sp];
  float acc = 0.0f;
  for (int j = 0; j < 9; j++) {
    const int h2 = hh + j / 3 - 1, w2 = ww + j % 3 - 1;
    const bool inb = (h2 >= 0 && h2 < 32 && w2 >= 0 && w2 < 32);
    const int sp2 = inb ? (h2 * 32 + w2) : 0;
    float dot = 0.0f;
    if (inb) {
      for (int i = threadIdx.x; i < 1536; i += 256) {
        const int hf = (i >= DMODEL) ? 1 : 0;
        const int c = i - hf * DMODEL;
        dot += cu[i] * x[((size_t)b * SEQ + hf * 1024 + sp2) * DMODEL + c];
      }
    }
    dot = block_reduce_sum(dot, sm);
    if (inb) {
      const float pn = norms[b * 1024 + sp2];
      acc += dot / (fmaxf(cn, 1e-8f) * fmaxf(pn, 1e-8f));
    }
  }
  if (threadIdx.x == 0) sim[blockIdx.x] = acc * (1.0f / 81.0f);
}

__global__ void scale_kernel(float* __restrict__ out, const float* __restrict__ sim) {
  const size_t idx = (size_t)blockIdx.x * 256 + threadIdx.x;
  const int b = (int)(idx / ((size_t)SEQ * DMODEL));
  const int rem = (int)(idx - (size_t)b * SEQ * DMODEL);
  const int n = rem / DMODEL;
  out[idx] *= sim[b * 1024 + (n & 1023)];
}

// ---------------------------------------------------------------------------
// Host orchestration
// ---------------------------------------------------------------------------
extern "C" void kernel_launch(void* const* d_in, const int* in_sizes, int n_in,
                              void* d_out, int out_size, void* d_ws, size_t ws_size,
                              hipStream_t stream) {
  (void)in_sizes; (void)n_in; (void)out_size; (void)ws_size;

  // Input order: setup_inputs() dict insertion order (recursive).
  const float* x     = (const float*)d_in[0];
  const float* e_wq  = (const float*)d_in[1];   const float* e_bq  = (const float*)d_in[2];
  const float* e_wk  = (const float*)d_in[3];   const float* e_bk  = (const float*)d_in[4];
  const float* e_wv  = (const float*)d_in[5];   const float* e_bv  = (const float*)d_in[6];
  const float* e_wp  = (const float*)d_in[7];   const float* e_bp  = (const float*)d_in[8];
  const float* e_w1  = (const float*)d_in[9];   const float* e_b1  = (const float*)d_in[10];
  const float* e_w2  = (const float*)d_in[11];  const float* e_b2  = (const float*)d_in[12];
  const float* e_g1  = (const float*)d_in[13];  const float* e_be1 = (const float*)d_in[14];
  const float* e_g2  = (const float*)d_in[15];  const float* e_be2 = (const float*)d_in[16];
  const float* proto = (const float*)d_in[17];
  const float* w_wq  = (const float*)d_in[18];  const float* w_bq  = (const float*)d_in[19];
  const float* w_wk  = (const float*)d_in[20];  const float* w_bk  = (const float*)d_in[21];
  const float* w_wv  = (const float*)d_in[22];  const float* w_bv  = (const float*)d_in[23];
  const float* w_wp  = (const float*)d_in[24];  const float* w_bp  = (const float*)d_in[25];
  const float* w_w1  = (const float*)d_in[26];  const float* w_b1  = (const float*)d_in[27];
  const float* w_w2  = (const float*)d_in[28];  const float* w_b2  = (const float*)d_in[29];
  const float* w_g1  = (const float*)d_in[30];  const float* w_be1 = (const float*)d_in[31];
  const float* w_g2  = (const float*)d_in[32];  const float* w_be2 = (const float*)d_in[33];

  char* wsp = (char*)d_ws;
  auto alloc = [&](size_t bytes) -> char* {
    char* p = wsp;
    wsp += (bytes + 255) & ~(size_t)255;
    return p;
  };

  const size_t BS = (size_t)BATCH * SEQ;

  // f16 transposed weights
  f16* e_wqT = (f16*)alloc((size_t)DMODEL * DMODEL * 2);
  f16* e_wkT = (f16*)alloc((size_t)DMODEL * DMODEL * 2);
  f16* e_wvT = (f16*)alloc((size_t)DMODEL * DMODEL * 2);
  f16* e_wpT = (f16*)alloc((size_t)DMODEL * DMODEL * 2);
  f16* e_w1T = (f16*)alloc((size_t)DMODEL * FFN * 2);
  f16* e_w2T = (f16*)alloc((size_t)FFN * DMODEL * 2);
  f16* d_wqT = (f16*)alloc((size_t)DMODEL * DMODEL * 2);
  f16* d_wkT = (f16*)alloc((size_t)DMODEL * DMODEL * 2);
  f16* d_wvT = (f16*)alloc((size_t)DMODEL * DMODEL * 2);
  f16* d_wpT = (f16*)alloc((size_t)DMODEL * DMODEL * 2);
  f16* d_w1T = (f16*)alloc((size_t)DMODEL * FFN * 2);
  f16* d_w2T = (f16*)alloc((size_t)FFN * DMODEL * 2);

  // activations
  f16*   lnH    = (f16*)alloc(BS * DMODEL * 2);
  f16*   protoH = (f16*)alloc((size_t)SEQ * DMODEL * 2);
  float* protoF = (float*)alloc((size_t)SEQ * DMODEL * 4);
  f16*   Qh     = (f16*)alloc(BS * DMODEL * 2);
  f16*   Kh     = (f16*)alloc(BS * DMODEL * 2);
  f16*   Vh     = (f16*)alloc(BS * DMODEL * 2);
  f16*   Oh     = (f16*)alloc(BS * DMODEL * 2);
  f16*   Hm     = (f16*)alloc(BS * FFN * 2);
  float* F1     = (float*)alloc(BS * DMODEL * 4);  // obj, then pe(f32)
  float* F2     = (float*)alloc(BS * DMODEL * 4);  // obj2, then out(pre-mlp)
  float* norms  = (float*)alloc((size_t)BATCH * 1024 * 4);
  float* simb   = (float*)alloc((size_t)BATCH * 1024 * 4);

  auto tr = [&](const float* W, f16* Wt, int K, int N) {
    const int total = K * N;
    transpose_f16<<<(total + 255) / 256, 256, 0, stream>>>(W, Wt, K, N);
  };
  tr(e_wq, e_wqT, DMODEL, DMODEL); tr(e_wk, e_wkT, DMODEL, DMODEL);
  tr(e_wv, e_wvT, DMODEL, DMODEL); tr(e_wp, e_wpT, DMODEL, DMODEL);
  tr(e_w1, e_w1T, DMODEL, FFN);    tr(e_w2, e_w2T, FFN, DMODEL);
  tr(w_wq, d_wqT, DMODEL, DMODEL); tr(w_wk, d_wkT, DMODEL, DMODEL);
  tr(w_wv, d_wvT, DMODEL, DMODEL); tr(w_wp, d_wpT, DMODEL, DMODEL);
  tr(w_w1, d_w1T, DMODEL, FFN);    tr(w_w2, d_w2T, FFN, DMODEL);

  auto gemm = [&](const f16* A, const f16* Bt, const float* bias, const float* res,
                  unsigned resMask, float* o32, f16* o16, int M, int N, int K,
                  float scale, int gelu) {
    dim3 grid(N / 128, M / 128);
    gemm_wmma<<<grid, 256, 0, stream>>>(A, Bt, bias, res, resMask, o32, o16, N, K,
                                        scale, gelu);
  };

  const float qscale = 0.125f;  // HEAD_DIM^-0.5

  // ------------------------- encoder -------------------------
  ln_kernel<<<(int)BS, 256, 0, stream>>>(x, e_g1, e_be1, lnH, nullptr);           // k = LN(x)
  ln_kernel<<<SEQ, 256, 0, stream>>>(proto, e_g1, e_be1, protoH, protoF);         // query
  gemm(protoH, e_wqT, e_bq, nullptr, 0u, nullptr, Qh, SEQ, DMODEL, DMODEL, qscale, 0);
  gemm(lnH, e_wkT, e_bk, nullptr, 0u, nullptr, Kh, (int)BS, DMODEL, DMODEL, 1.0f, 0);
  gemm(lnH, e_wvT, e_bv, nullptr, 0u, nullptr, Vh, (int)BS, DMODEL, DMODEL, 1.0f, 0);
  attn_kernel<<<dim3(SEQ / 128, 12, BATCH), 256, 0, stream>>>(Qh, 0L, Kh, Vh, Oh);
  gemm(Oh, e_wpT, e_bp, protoF, (unsigned)(SEQ - 1), F1, nullptr,
       (int)BS, DMODEL, DMODEL, 1.0f, 0);                                          // obj
  ln_kernel<<<(int)BS, 256, 0, stream>>>(F1, e_g2, e_be2, lnH, nullptr);
  gemm(lnH, e_w1T, e_b1, nullptr, 0u, nullptr, Hm, (int)BS, FFN, DMODEL, 1.0f, 1); // gelu
  gemm(Hm, e_w2T, e_b2, F1, 0xFFFFFFFFu, F2, nullptr, (int)BS, DMODEL, FFN, 1.0f, 0); // obj2

  // ------------------------- decoder -------------------------
  ln_kernel<<<(int)BS, 256, 0, stream>>>(x, w_g1, w_be1, lnH, F1);                 // pe
  gemm(lnH, d_wqT, w_bq, nullptr, 0u, nullptr, Qh, (int)BS, DMODEL, DMODEL, qscale, 0);
  ln_kernel<<<(int)BS, 256, 0, stream>>>(F2, w_g1, w_be1, lnH, nullptr);           // of
  gemm(lnH, d_wkT, w_bk, nullptr, 0u, nullptr, Kh, (int)BS, DMODEL, DMODEL, 1.0f, 0);
  gemm(lnH, d_wvT, w_bv, nullptr, 0u, nullptr, Vh, (int)BS, DMODEL, DMODEL, 1.0f, 0);
  attn_kernel<<<dim3(SEQ / 128, 12, BATCH), 256, 0, stream>>>(
      Qh, (long)SEQ * DMODEL, Kh, Vh, Oh);
  gemm(Oh, d_wpT, w_bp, F1, 0xFFFFFFFFu, F2, nullptr, (int)BS, DMODEL, DMODEL, 1.0f, 0);
  ln_kernel<<<(int)BS, 256, 0, stream>>>(F2, w_g2, w_be2, lnH, nullptr);
  gemm(lnH, d_w1T, w_b1, nullptr, 0u, nullptr, Hm, (int)BS, FFN, DMODEL, 1.0f, 1);
  gemm(Hm, d_w2T, w_b2, F2, 0xFFFFFFFFu, (float*)d_out, nullptr,
       (int)BS, DMODEL, FFN, 1.0f, 0);                                             // final residual

  // ------------------------- bcim -------------------------
  norm_kernel<<<BATCH * 1024, 256, 0, stream>>>((const float*)d_out, norms);
  sim_kernel<<<BATCH * 1024, 256, 0, stream>>>((const float*)d_out, norms, simb);
  scale_kernel<<<(int)(BS * DMODEL / 256), 256, 0, stream>>>((float*)d_out, simb);
}
